// Model_82935818486477
// MI455X (gfx1250) — matrix operand order repaired and checked by reading, not compiled, not run on hardware
//
#include <hip/hip_runtime.h>
#include <hip/hip_bf16.h>

typedef __attribute__((ext_vector_type(16))) _Float16 v16h;
typedef __attribute__((ext_vector_type(8)))  float    v8f;
typedef __attribute__((ext_vector_type(4)))  float    v4f;

#define T_LEN  512
#define B_SZ   128
#define E_SZ   128
#define H_SZ   256
#define K_TAGS 10
#define GATES  1024          // 4*H
#define KDIM   384           // E + H  (fused [x | h] GEMM)
#define KBLK   12            // KDIM / 32

// Branchless fast nonlinearities (single v_exp_f32 + v_rcp_f32, no exec-mask
// branches in the serial recurrence; saturate correctly at the extremes).
__device__ __forceinline__ float sigmoid_fast(float x) {
  return __builtin_amdgcn_rcpf(1.f + __expf(-x));
}
__device__ __forceinline__ float tanh_fast(float x) {
  return 2.f * __builtin_amdgcn_rcpf(1.f + __expf(-2.f * x)) - 1.f;
}

// ---------------------------------------------------------------------------
// Pre-swizzle combined B matrix  B[k][n] = (k<E) ? w_ih[n][k] : w_hh[n][k-E]
// into WMMA B-fragment layout: [colTile(64)][kBlock(12)][lane(32)][half(16)]
// so each lane's 16 halves are contiguous (two global_load_b128 per fragment).
// ---------------------------------------------------------------------------
__global__ void prep_wfrag_kernel(const float* __restrict__ w_ih,
                                  const float* __restrict__ w_hh,
                                  _Float16* __restrict__ wfrag) {
  int idx = blockIdx.x * blockDim.x + threadIdx.x;   // over KDIM*GATES
  if (idx >= KDIM * GATES) return;
  int k = idx / GATES;
  int n = idx % GATES;
  float v = (k < E_SZ) ? w_ih[n * E_SZ + k] : w_hh[n * H_SZ + (k - E_SZ)];
  int ct  = n >> 4, nl  = n & 15;
  int kb  = k >> 5, k32 = k & 31;
  int lane = nl + ((k32 & 16) ? 16 : 0);
  int hh   = k32 & 15;
  wfrag[(((ct * KBLK + kb) * 32 + lane) << 4) + hh] = (_Float16)v;
}

// ---------------------------------------------------------------------------
// Persistent LSTM kernel: blockIdx.x = 0 forward, 1 backward.
// 1024 threads = 32 waves on one WGP. Each wave owns 4 (rowTile, jTile)
// pairs; per pair it accumulates the 4 gate tiles with v_wmma.
// VGPR-pressure design: cell state c lives in LDS (f32), h is double-
// buffered in LDS so each pair's h tile is stored immediately after its
// cell update -> no long-lived hnew registers, 2 barriers per step, and
// no scratch spills in the 512-step serial chain.
// ---------------------------------------------------------------------------
__global__ void __launch_bounds__(1024) lstm_kernel(
    const int*      __restrict__ tokens,   // (B,T)
    const float*    __restrict__ embed,    // (V,E)
    const _Float16* __restrict__ wfragF,
    const _Float16* __restrict__ wfragB,
    const float*    __restrict__ biasF,    // (4H,)
    const float*    __restrict__ biasB,
    const float*    __restrict__ lin_w,    // (K, 2H)
    float*          __restrict__ ebufF,    // (T,B,K)
    float*          __restrict__ ebufB) {
  extern __shared__ char smem[];
  size_t so = 0;
  _Float16* xs  = (_Float16*)(smem + so); so += (size_t)B_SZ * E_SZ * 2;  // 32KB
  _Float16* hs0 = (_Float16*)(smem + so); so += (size_t)B_SZ * H_SZ * 2;  // 64KB
  _Float16* hs1 = (_Float16*)(smem + so); so += (size_t)B_SZ * H_SZ * 2;  // 64KB
  float*    cs  = (float*)(smem + so);    so += (size_t)B_SZ * H_SZ * 4;  // 128KB
  float*    we  = (float*)(smem + so);                                     // 10KB

  const int dir = blockIdx.x;
  const _Float16* wfrag = dir ? wfragB : wfragF;
  const float*    bias  = dir ? biasB  : biasF;
  float*          ebuf  = dir ? ebufB  : ebufF;

  const int tid  = threadIdx.x;
  const int lane = tid & 31;
  const int wave = tid >> 5;
  const int rT   = wave >> 2;           // row tile 0..7   (16 rows of B)
  const int jt0  = (wave & 3) * 4;      // first of 4 hidden col tiles
  const int mrow = lane & 15;           // A row within tile
  const int hi   = lane >> 4;           // lane half select
  const int jcol = lane & 15;           // C/D column within tile

  // init h = 0 (read buffer of step 0), c = 0, stage emission weights
  for (int i = tid; i < B_SZ * H_SZ; i += 1024) {
    hs0[i] = (_Float16)0.f;
    cs[i]  = 0.f;
  }
  for (int i = tid; i < K_TAGS * H_SZ; i += 1024) {
    int k = i >> 8, j = i & 255;
    we[i] = lin_w[k * (2 * H_SZ) + dir * H_SZ + j];
  }

  // per-(pair,gate) bias, constant across steps (col = g*256 + jt*16 + jcol);
  // bias is constant across the 8 tile rows, so it seeds the accumulators.
  float bReg[4][4];
#pragma unroll
  for (int p = 0; p < 4; ++p)
#pragma unroll
    for (int g = 0; g < 4; ++g)
      bReg[p][g] = bias[g * H_SZ + (jt0 + p) * 16 + jcol];

  __syncthreads();

  for (int step = 0; step < T_LEN; ++step) {
    const int tt = dir ? (T_LEN - 1 - step) : step;
    _Float16* hrd = (step & 1) ? hs1 : hs0;
    _Float16* hwr = (step & 1) ? hs0 : hs1;

    // gather x_t = embed[tokens[:, tt]] into LDS (f32 -> f16)
    for (int i = tid; i < B_SZ * E_SZ; i += 1024) {
      int b = i >> 7, e = i & 127;
      int tok = tokens[b * T_LEN + tt];
      xs[i] = (_Float16)embed[tok * E_SZ + e];
    }
    __syncthreads();   // B1: xs ready, hrd complete (from previous step)

#pragma unroll
    for (int p = 0; p < 4; ++p) {
      const int jt = jt0 + p;
      v8f acc[4];
#pragma unroll
      for (int g = 0; g < 4; ++g)
#pragma unroll
        for (int d = 0; d < 8; ++d) acc[g][d] = bReg[p][g];

      for (int kb = 0; kb < KBLK; ++kb) {
        // A fragment (16x32 f16) from LDS [x | h_rd], contiguous k = 16*hi + hh (same as wfrag)
        v16h a;
#pragma unroll
        for (int hh = 0; hh < 16; ++hh) {
          int kl = hi * 16 + hh;
          int k  = kb * 32 + kl;
          int row = rT * 16 + mrow;
          _Float16 val = (k < E_SZ) ? xs[row * E_SZ + k]
                                    : hrd[row * H_SZ + (k - E_SZ)];
          a[hh] = val;
        }
        // 4 gate tiles share the A fragment
#pragma unroll
        for (int g = 0; g < 4; ++g) {
          int ct = g * 16 + jt;
          const v16h bfr =
              *(const v16h*)(wfrag + (((ct * KBLK + kb) * 32 + lane) << 4));
          acc[g] = __builtin_amdgcn_wmma_f32_16x16x32_f16(
              false, a, false, bfr, (short)0, acc[g], false, false);
        }
      }

      // LSTM cell update in accumulator layout (gate order i,f,g,o);
      // c lives in LDS, h tile stored immediately to the write buffer.
      const int colB = jt * 16 + jcol;
#pragma unroll
      for (int d = 0; d < 8; ++d) {
        const int row = rT * 16 + d + hi * 8;
        const int ci  = row * H_SZ + colB;
        float c = sigmoid_fast(acc[1][d]) * cs[ci] +
                  sigmoid_fast(acc[0][d]) * tanh_fast(acc[2][d]);
        cs[ci] = c;
        hwr[ci] = (_Float16)(sigmoid_fast(acc[3][d]) * tanh_fast(c));
      }
    }
    __syncthreads();   // B2: h_t (hwr) complete for all tiles

    // fused emission projection: e[b][k] = sum_j h[b][j] * We[k][j]
    // vectorized: h as v16h (ds_load_b128), We as float4.
    {
      int b = tid & 127;
      int k = tid >> 7;                      // 0..7
      const v16h* hrow = (const v16h*)(hwr + b * H_SZ);
      {
        const v4f* wrow = (const v4f*)(we + k * H_SZ);
        float accE = 0.f;
#pragma unroll
        for (int v = 0; v < 16; ++v) {
          v16h hv = hrow[v];
#pragma unroll
          for (int q = 0; q < 4; ++q) {
            v4f wq = wrow[v * 4 + q];
#pragma unroll
            for (int u = 0; u < 4; ++u)
              accE += (float)hv[q * 4 + u] * wq[u];
          }
        }
        ebuf[((size_t)tt * B_SZ + b) * K_TAGS + k] = accE;
      }
      if (tid < 256) {
        int k2 = 8 + (tid >> 7);             // 8,9
        const v4f* wrow = (const v4f*)(we + k2 * H_SZ);
        float accE = 0.f;
#pragma unroll
        for (int v = 0; v < 16; ++v) {
          v16h hv = hrow[v];
#pragma unroll
          for (int q = 0; q < 4; ++q) {
            v4f wq = wrow[v * 4 + q];
#pragma unroll
            for (int u = 0; u < 4; ++u)
              accE += (float)hv[q * 4 + u] * wq[u];
          }
        }
        ebuf[((size_t)tt * B_SZ + b) * K_TAGS + k2] = accE;
      }
    }
    // no barrier needed here: next step writes the other h buffer, and B1
    // orders the next xs fill against everything.
  }
}

// ---------------------------------------------------------------------------
// CRF: numerator + forward algorithm (K=10). One lane per batch element.
// em(t,b,k) = ebufF + ebufB + lin_b reconstructed on the fly.
// ---------------------------------------------------------------------------
__global__ void __launch_bounds__(128) crf_kernel(
    const int*           __restrict__ target,  // (B,T)
    const unsigned char* __restrict__ mask,    // (B,T) bool
    const float* __restrict__ startv, const float* __restrict__ endv,
    const float* __restrict__ trans,  const float* __restrict__ lin_b,
    const float* __restrict__ ebufF,  const float* __restrict__ ebufB,
    float* __restrict__ out) {
  __shared__ float transL[K_TAGS * K_TAGS];
  __shared__ float red[128];
  const int tid = threadIdx.x;
  for (int i = tid; i < K_TAGS * K_TAGS; i += 128) transL[i] = trans[i];
  __syncthreads();

  const int b = tid;
  float lb[K_TAGS], st[K_TAGS], en[K_TAGS];
#pragma unroll
  for (int k = 0; k < K_TAGS; ++k) {
    lb[k] = lin_b[k]; st[k] = startv[k]; en[k] = endv[k];
  }
  auto em = [&](int t, int k) -> float {
    size_t o = ((size_t)t * B_SZ + b) * K_TAGS + k;
    return ebufF[o] + ebufB[o] + lb[k];
  };

  int   tag_prev = target[b * T_LEN + 0];
  float num      = st[tag_prev] + em(0, tag_prev);
  int   seq_len  = mask[b * T_LEN + 0] ? 1 : 0;

  float alpha[K_TAGS];
#pragma unroll
  for (int k = 0; k < K_TAGS; ++k) alpha[k] = st[k] + em(0, k);

  for (int t = 1; t < T_LEN; ++t) {
    const bool m   = mask[b * T_LEN + t] != 0;
    const int  tag = target[b * T_LEN + t];
    if (m) num += em(t, tag) + transL[tag_prev * K_TAGS + tag];
    tag_prev = tag;
    seq_len += m ? 1 : 0;

    float nxt[K_TAGS];
#pragma unroll
    for (int k2 = 0; k2 < K_TAGS; ++k2) {
      float mx = -1e30f;
#pragma unroll
      for (int k1 = 0; k1 < K_TAGS; ++k1)
        mx = fmaxf(mx, alpha[k1] + transL[k1 * K_TAGS + k2]);
      float s = 0.f;
#pragma unroll
      for (int k1 = 0; k1 < K_TAGS; ++k1)
        s += __expf(alpha[k1] + transL[k1 * K_TAGS + k2] - mx);
      nxt[k2] = mx + __logf(s) + em(t, k2);
    }
#pragma unroll
    for (int k2 = 0; k2 < K_TAGS; ++k2) alpha[k2] = m ? nxt[k2] : alpha[k2];
  }

  int last_tag = target[b * T_LEN + (seq_len - 1)];
  num += en[last_tag];

  float mx = -1e30f;
#pragma unroll
  for (int k = 0; k < K_TAGS; ++k) mx = fmaxf(mx, alpha[k] + en[k]);
  float s = 0.f;
#pragma unroll
  for (int k = 0; k < K_TAGS; ++k) s += __expf(alpha[k] + en[k] - mx);
  float denom = mx + __logf(s);

  red[tid] = num - denom;
  __syncthreads();
  for (int stp = 64; stp > 0; stp >>= 1) {
    if (tid < stp) red[tid] += red[tid + stp];
    __syncthreads();
  }
  if (tid == 0) out[0] = -(red[0] / (float)B_SZ);
}

// ---------------------------------------------------------------------------
extern "C" void kernel_launch(void* const* d_in, const int* in_sizes, int n_in,
                              void* d_out, int out_size, void* d_ws,
                              size_t ws_size, hipStream_t stream) {
  const int*           tokens = (const int*)d_in[0];
  const int*           target = (const int*)d_in[1];
  const unsigned char* mask   = (const unsigned char*)d_in[2];
  const float*         embed  = (const float*)d_in[3];
  const float*         w_ih_f = (const float*)d_in[4];
  const float*         w_hh_f = (const float*)d_in[5];
  const float*         b_f    = (const float*)d_in[6];
  const float*         w_ih_b = (const float*)d_in[7];
  const float*         w_hh_b = (const float*)d_in[8];
  const float*         b_b    = (const float*)d_in[9];
  const float*         lin_w  = (const float*)d_in[10];
  const float*         lin_b  = (const float*)d_in[11];
  const float*         startv = (const float*)d_in[12];
  const float*         endv   = (const float*)d_in[13];
  const float*         trans  = (const float*)d_in[14];

  char* ws = (char*)d_ws;
  size_t o = 0;
  _Float16* wfragF = (_Float16*)(ws + o); o += (size_t)KDIM * GATES * 2;
  _Float16* wfragB = (_Float16*)(ws + o); o += (size_t)KDIM * GATES * 2;
  float*    ebufF  = (float*)(ws + o);    o += (size_t)T_LEN * B_SZ * K_TAGS * 4;
  float*    ebufB  = (float*)(ws + o);    o += (size_t)T_LEN * B_SZ * K_TAGS * 4;
  (void)ws_size; (void)in_sizes; (void)n_in; (void)out_size;

  const int nfrag = KDIM * GATES;
  prep_wfrag_kernel<<<(nfrag + 255) / 256, 256, 0, stream>>>(w_ih_f, w_hh_f, wfragF);
  prep_wfrag_kernel<<<(nfrag + 255) / 256, 256, 0, stream>>>(w_ih_b, w_hh_b, wfragB);

  const size_t shmem = (size_t)B_SZ * E_SZ * 2        // xs
                     + (size_t)B_SZ * H_SZ * 2 * 2    // hs double buffer
                     + (size_t)B_SZ * H_SZ * 4        // c state (f32)
                     + (size_t)K_TAGS * H_SZ * 4;     // We
  lstm_kernel<<<2, 1024, shmem, stream>>>(tokens, embed, wfragF, wfragB,
                                          b_f, b_b, lin_w, ebufF, ebufB);

  crf_kernel<<<1, 128, 0, stream>>>(target, mask, startv, endv, trans, lin_b,
                                    ebufF, ebufB, (float*)d_out);
}
